// QuantLinear_57878979281491
// MI455X (gfx1250) — 8-bit matrix operand order repaired and checked by reading, compiled, not run on hardware
//
#include <hip/hip_runtime.h>
#include <hip/hip_bf16.h>

typedef int   v8i __attribute__((ext_vector_type(8)));
typedef float v8f __attribute__((ext_vector_type(8)));

#define TOKENS 8192
#define IN_F   4096
#define OUT_F  4096
#define GROUP  128
#define NGROUP (IN_F / GROUP)

#define WMMA_IU8(a, b, c) \
  __builtin_amdgcn_wmma_i32_16x16x64_iu8(true, (a), true, (b), (c), false, false)


__device__ __forceinline__ float wave_max32(float v) {
#pragma unroll
  for (int off = 16; off >= 1; off >>= 1)
    v = fmaxf(v, __shfl_xor(v, off, 32));
  return v;
}

__device__ __forceinline__ int quant_pack4(float4 v, float inv) {
  int q0 = (int)fminf(fmaxf(rintf(v.x * inv), -128.f), 127.f);
  int q1 = (int)fminf(fmaxf(rintf(v.y * inv), -128.f), 127.f);
  int q2 = (int)fminf(fmaxf(rintf(v.z * inv), -128.f), 127.f);
  int q3 = (int)fminf(fmaxf(rintf(v.w * inv), -128.f), 127.f);
  return (q0 & 255) | ((q1 & 255) << 8) | ((q2 & 255) << 16) | ((q3 & 255) << 24);
}

__global__ void wquant_kernel(const float* __restrict__ w,
                              signed char* __restrict__ wq,
                              float* __restrict__ wsc) {
  const int row  = blockIdx.x;
  const int t    = threadIdx.x;
  const int lane = t & 31;
  const int wv   = t >> 5;
  const float* wr = w + (size_t)row * IN_F;

  float4 v[4];
  float m = 0.f;
#pragma unroll
  for (int i = 0; i < 4; ++i) {
    v[i] = *(const float4*)(wr + t * 16 + i * 4);
    m = fmaxf(m, fmaxf(fmaxf(fabsf(v[i].x), fabsf(v[i].y)),
                       fmaxf(fabsf(v[i].z), fabsf(v[i].w))));
  }
  m = wave_max32(m);
  __shared__ float red[8];
  if (lane == 0) red[wv] = m;
  __syncthreads();
  float mm = red[0];
#pragma unroll
  for (int i = 1; i < 8; ++i) mm = fmaxf(mm, red[i]);

  const float scale = fmaxf(mm * (1.f / 127.f), 1e-8f);
  if (t == 0) wsc[row] = scale;
  const float inv = 1.f / scale;

  int4 packed;
  packed.x = quant_pack4(v[0], inv);
  packed.y = quant_pack4(v[1], inv);
  packed.z = quant_pack4(v[2], inv);
  packed.w = quant_pack4(v[3], inv);
  *(int4*)(wq + (size_t)row * IN_F + t * 16) = packed;
}

__global__ void xquant_kernel(const float* __restrict__ x,
                              signed char* __restrict__ xq,
                              float* __restrict__ xs) {
  const int tok   = blockIdx.x * 2 + (threadIdx.x >> 7);
  const int i     = threadIdx.x & 127;
  const int chunk = i >> 1;
  const int half  = i & 1;
  const int g     = i >> 2;

  const float* xr = x + (size_t)tok * IN_F + chunk * 64 + half * 16;

  float4 p[8];
  float m = 0.f;
#pragma unroll
  for (int piece = 0; piece < 4; ++piece) {
    p[piece * 2]     = *(const float4*)(xr + (piece & 1) * 8 + (piece >> 1) * 32);
    p[piece * 2 + 1] = *(const float4*)(xr + (piece & 1) * 8 + (piece >> 1) * 32 + 4);
    float4 a = p[piece * 2], b = p[piece * 2 + 1];
    m = fmaxf(m, fmaxf(fmaxf(fabsf(a.x), fabsf(a.y)), fmaxf(fabsf(a.z), fabsf(a.w))));
    m = fmaxf(m, fmaxf(fmaxf(fabsf(b.x), fabsf(b.y)), fmaxf(fabsf(b.z), fabsf(b.w))));
  }
  m = fmaxf(m, __shfl_xor(m, 1, 32));
  m = fmaxf(m, __shfl_xor(m, 2, 32));

  const float scale = fmaxf(m * (1.f / 127.f), 1e-8f);
  const float inv = 1.f / scale;
  if ((i & 3) == 0) xs[(size_t)g * TOKENS + tok] = scale;

  int q[8];
#pragma unroll
  for (int d = 0; d < 8; ++d) q[d] = quant_pack4(p[d], inv);

  signed char* dst = xq +
      ((((size_t)(tok >> 4) * 64 + chunk) * 2 + half) * 16 + (tok & 15)) * 32;
  *(int4*)(dst)      = make_int4(q[0], q[1], q[2], q[3]);
  *(int4*)(dst + 16) = make_int4(q[4], q[5], q[6], q[7]);
}

__device__ __forceinline__ v8i load_a_frag(const signed char* p) {
  int4 q0 = *(const int4*)(p);
  int4 q1 = *(const int4*)(p + 16);
  v8i r;
  r[0] = q0.x; r[1] = q0.y; r[2] = q0.z; r[3] = q0.w;
  r[4] = q1.x; r[5] = q1.y; r[6] = q1.z; r[7] = q1.w;
  return r;
}

__device__ __forceinline__ v8i load_b_frag(const signed char* p) {
  int4 q0 = *(const int4*)(p);
  int4 q1 = *(const int4*)(p + 32);
  v8i r;
  r[0] = q0.x; r[1] = q0.y; r[2] = q0.z; r[3] = q0.w;
  r[4] = q1.x; r[5] = q1.y; r[6] = q1.z; r[7] = q1.w;
  return r;
}

struct GroupAcc {
  v8i i[2][2];
  v8f sx[2];
};

__global__ void __launch_bounds__(128, 2)
gemm_kernel(const signed char* __restrict__ xq,
            const float* __restrict__ xs,
            const signed char* __restrict__ wq,
            const float* __restrict__ wsc,
            const float* __restrict__ bias,
            float* __restrict__ out) {
  const int lane = threadIdx.x & 31;
  const int wv   = threadIdx.x >> 5;
  const int l16  = lane & 15;
  const int hi   = lane >> 4;
  const int wy   = wv >> 1;
  const int wx   = wv & 1;
  const int tb   = blockIdx.y * 64 + wy * 32;
  const int ob   = blockIdx.x * 64 + wx * 32;

  v8f facc[2][2];

  const signed char* aBase = xq + (size_t)(tb >> 4) * 65536 +
                             (size_t)hi * 512 + (size_t)l16 * 32;
  const signed char* bBase0 = wq + (size_t)(ob + l16) * IN_F + hi * 16;
  const signed char* bBase1 = bBase0 + (size_t)16 * IN_F;

  auto compute_group = [&](int g, GroupAcc& acc) {
    const size_t ga = (size_t)g * 2048;
    const int    kb = g * GROUP;

    {
      v8i a0 = load_a_frag(aBase + ga);
      v8i a1 = load_a_frag(aBase + 65536 + ga);
      v8i b0 = load_b_frag(bBase0 + kb);
      v8i b1 = load_b_frag(bBase1 + kb);
      const v8i z = 0;
      acc.i[0][0] = WMMA_IU8(a0, b0, z);
      acc.i[0][1] = WMMA_IU8(a0, b1, z);
      acc.i[1][0] = WMMA_IU8(a1, b0, z);
      acc.i[1][1] = WMMA_IU8(a1, b1, z);
    }
    {
      v8i a0 = load_a_frag(aBase + ga + 1024);
      v8i a1 = load_a_frag(aBase + 65536 + ga + 1024);
      v8i b0 = load_b_frag(bBase0 + kb + 64);
      v8i b1 = load_b_frag(bBase1 + kb + 64);
      acc.i[0][0] = WMMA_IU8(a0, b0, acc.i[0][0]);
      acc.i[0][1] = WMMA_IU8(a0, b1, acc.i[0][1]);
      acc.i[1][0] = WMMA_IU8(a1, b0, acc.i[1][0]);
      acc.i[1][1] = WMMA_IU8(a1, b1, acc.i[1][1]);
    }

    const float* sp = xs + (size_t)g * TOKENS + tb + hi * 8;
#pragma unroll
    for (int r = 0; r < 2; ++r) {
      float4 s0 = *(const float4*)(sp + r * 16);
      float4 s1 = *(const float4*)(sp + r * 16 + 4);
      acc.sx[r][0] = s0.x; acc.sx[r][1] = s0.y;
      acc.sx[r][2] = s0.z; acc.sx[r][3] = s0.w;
      acc.sx[r][4] = s1.x; acc.sx[r][5] = s1.y;
      acc.sx[r][6] = s1.z; acc.sx[r][7] = s1.w;
    }
  };

  auto rescale_first = [&](GroupAcc& acc) {
#pragma unroll
    for (int r = 0; r < 2; ++r)
#pragma unroll
      for (int c = 0; c < 2; ++c)
#pragma unroll
        for (int v = 0; v < 8; ++v)
          facc[r][c][v] = (float)acc.i[r][c][v] * acc.sx[r][v];
  };
  auto rescale = [&](GroupAcc& acc) {
#pragma unroll
    for (int r = 0; r < 2; ++r)
#pragma unroll
      for (int c = 0; c < 2; ++c)
#pragma unroll
        for (int v = 0; v < 8; ++v)
          facc[r][c][v] += (float)acc.i[r][c][v] * acc.sx[r][v];
  };

  GroupAcc accA, accB;
  compute_group(0, accA);
  compute_group(1, accB);
  rescale_first(accA);
#pragma unroll 3
  for (int g = 2; g < NGROUP; g += 2) {
    compute_group(g, accA);
    rescale(accB);
    compute_group(g + 1, accB);
    rescale(accA);
  }
  rescale(accB);

#pragma unroll
  for (int c = 0; c < 2; ++c) {
    const int col = ob + c * 16 + l16;
    const float swv = wsc[col];
    const float bv  = bias[col];
#pragma unroll
    for (int r = 0; r < 2; ++r)
#pragma unroll
      for (int v = 0; v < 8; ++v) {
        const int tok = tb + r * 16 + hi * 8 + v;
        out[(size_t)tok * OUT_F + col] = facc[r][c][v] * swv + bv;
      }
  }
}

extern "C" void kernel_launch(void* const* d_in, const int* in_sizes, int n_in,
                              void* d_out, int out_size, void* d_ws, size_t ws_size,
                              hipStream_t stream) {
  const float* x    = (const float*)d_in[0];
  const float* w    = (const float*)d_in[1];
  const float* bias = (const float*)d_in[2];
  float* out = (float*)d_out;

  char* ws = (char*)d_ws;
  size_t off = 0;
  signed char* wq = (signed char*)(ws + off); off += (size_t)OUT_F * IN_F;
  signed char* xq = (signed char*)(ws + off); off += (size_t)TOKENS * IN_F;
  float* wsc = (float*)(ws + off);            off += (size_t)OUT_F * sizeof(float);
  float* xs  = (float*)(ws + off);

  wquant_kernel<<<OUT_F, 256, 0, stream>>>(w, wq, wsc);
  xquant_kernel<<<TOKENS / 2, 256, 0, stream>>>(x, xq, xs);

  dim3 grid(OUT_F / 64, TOKENS / 64);
  gemm_kernel<<<grid, 128, 0, stream>>>(xq, xs, wq, wsc, bias, out);
}
